// TopoFlowAttention_79370995631051
// MI455X (gfx1250) — compile-verified, matrix operand order checked by reading, not run on hardware
//
#include <hip/hip_runtime.h>
#include <hip/hip_bf16.h>

typedef __bf16 bf16_t;
typedef __attribute__((ext_vector_type(16))) __bf16 v16bf;
typedef __attribute__((ext_vector_type(8)))  __bf16 v8bf;
typedef __attribute__((ext_vector_type(8)))  float  v8f;

// ---------------------------------------------------------------------------
// WMMA helpers (CDNA5 wave32, 16x16x32 bf16 -> f32)
// ---------------------------------------------------------------------------
__device__ __forceinline__ v8f wmma_bf16(v16bf a, v16bf b, v8f c) {
  return __builtin_amdgcn_wmma_f32_16x16x32_bf16(
      /*neg_a=*/false, a, /*neg_b=*/false, b,
      /*c_mod=*/(short)0, c, /*reuse_a=*/false, /*reuse_b=*/false);
}

// Load a 16x32 bf16 fragment (A-style; B-style uses a [N][K] row-major source).
// Lane L: row = L%16, contiguous 8 elems at kb=(L/16)*8 and at kb+16.
__device__ __forceinline__ v16bf load_frag(const bf16_t* __restrict__ p, int stride) {
  const int lane = threadIdx.x & 31;
  const int r  = lane & 15;
  const int kb = (lane >> 4) << 3;
  const bf16_t* row = p + r * stride + kb;
  v8bf lo = *(const v8bf*)(row);
  v8bf hi = *(const v8bf*)(row + 16);
  v16bf f;
#pragma unroll
  for (int i = 0; i < 8; ++i) { f[i] = lo[i]; f[i + 8] = hi[i]; }
  return f;
}

// ---------------------------------------------------------------------------
// Kernel 0: f32 -> bf16 conversion
// ---------------------------------------------------------------------------
__global__ void cvt_f32_to_bf16(const float* __restrict__ src,
                                bf16_t* __restrict__ dst, int n) {
  int i = blockIdx.x * blockDim.x + threadIdx.x;
  if (i < n) dst[i] = (bf16_t)src[i];
}

// ---------------------------------------------------------------------------
// GEMM core: one wave computes a 16x64 tile of  A[16xK] @ B^T (B is [N][K]).
// Fully unrolled ping-pong double buffering: phase X's fragment loads are
// issued before phase Y's WMMAs; no register rotation copies survive.
// K = 768 (24 k-steps of 32 -> 12 double-steps).
// ---------------------------------------------------------------------------
__device__ __forceinline__ void gemm_16x64_k768(const bf16_t* __restrict__ aptr,
                                                const bf16_t* __restrict__ bptr,
                                                v8f acc[4]) {
  v16bf aA, bA[4], aB, bB[4];
  aA = load_frag(aptr, 768);
#pragma unroll
  for (int c = 0; c < 4; ++c) bA[c] = load_frag(bptr + c * 16 * 768, 768);

#pragma unroll
  for (int k0 = 0; k0 < 768; k0 += 64) {
    // phase-B loads (k0+32 <= 736, always valid)
    aB = load_frag(aptr + k0 + 32, 768);
#pragma unroll
    for (int c = 0; c < 4; ++c)
      bB[c] = load_frag(bptr + c * 16 * 768 + k0 + 32, 768);
    // phase-A WMMAs
#pragma unroll
    for (int c = 0; c < 4; ++c) acc[c] = wmma_bf16(aA, bA[c], acc[c]);
    // phase-A loads for next double-step (compile-time guard: full unroll)
    if (k0 + 64 < 768) {
      aA = load_frag(aptr + k0 + 64, 768);
#pragma unroll
      for (int c = 0; c < 4; ++c)
        bA[c] = load_frag(bptr + c * 16 * 768 + k0 + 64, 768);
    }
    // phase-B WMMAs
#pragma unroll
    for (int c = 0; c < 4; ++c) acc[c] = wmma_bf16(aB, bB[c], acc[c]);
  }
}

// ---------------------------------------------------------------------------
// Kernel 1: QKV GEMM  Y = x @ Wqkv^T   (M=4096, K=768, N=2304)
// Outputs scattered into q [b,h,i,d]*0.125, k [b,h,j,d], vT [b,h,d,j] (bf16).
// ---------------------------------------------------------------------------
__global__ void __launch_bounds__(256)
qkv_gemm(const bf16_t* __restrict__ xb, const bf16_t* __restrict__ wb,
         bf16_t* __restrict__ q, bf16_t* __restrict__ kmat,
         bf16_t* __restrict__ vT) {
  const int lane = threadIdx.x & 31;
  const int wave = blockIdx.x * (blockDim.x >> 5) + (threadIdx.x >> 5);
  const int row_t = wave / 36;        // 256 row tiles of 16
  const int col_g = wave % 36;        // 36 col groups of 64 (N=2304)
  const int i0 = row_t * 16;
  const int n0 = col_g * 64;

  v8f acc[4] = {};
  gemm_16x64_k768(xb + i0 * 768, wb + n0 * 768, acc);

  const int half = lane >> 4, col = lane & 15;
#pragma unroll
  for (int c = 0; c < 4; ++c) {
    const int n = n0 + c * 16 + col;
    const int s = n / 768;            // 0=q 1=k 2=v
    const int r = n % 768;
    const int h = r >> 6;
    const int d = r & 63;
#pragma unroll
    for (int v = 0; v < 8; ++v) {
      const int ig = i0 + v + half * 8;      // global row in [0,4096)
      const int b  = ig >> 10;
      const int il = ig & 1023;
      const float val = acc[c][v];
      if (s == 0) {
        q[(((b * 12 + h) << 10) + il) * 64 + d] = (bf16_t)(val * 0.125f);
      } else if (s == 1) {
        kmat[(((b * 12 + h) << 10) + il) * 64 + d] = (bf16_t)val;
      } else {
        vT[(((b * 12 + h) << 6) + d) * 1024 + il] = (bf16_t)val;
      }
    }
  }
}

// ---------------------------------------------------------------------------
// Kernel 2: flash-style attention. One wave owns 16 query rows of one (b,h).
// Pipelined inner loop: S-WMMAs -> prefetch next K tile + load V tile ->
// softmax (covers the loads) -> P via LDS -> PV-WMMAs.
// ---------------------------------------------------------------------------
__global__ void __launch_bounds__(128)
attn_kernel(const bf16_t* __restrict__ q, const bf16_t* __restrict__ kmat,
            const bf16_t* __restrict__ vT, const float* __restrict__ elev,
            const float* __restrict__ alpha_p, bf16_t* __restrict__ att) {
  __shared__ bf16_t ldsP[4][16 * 32];

  const int lane = threadIdx.x & 31;
  const int w    = threadIdx.x >> 5;
  const int wave = blockIdx.x * 4 + w;        // 3072 waves total
  const int it = wave & 63;                   // 64 i-tiles
  const int h  = (wave >> 6) % 12;
  const int b  = wave / 768;
  const int i0 = it * 16;

  const float alpha = *alpha_p;
  const int half = lane >> 4, col = lane & 15;
  const float* eb = elev + b * 1024;

  float e_i[8];
#pragma unroll
  for (int v = 0; v < 8; ++v) e_i[v] = eb[i0 + v + half * 8];

  const bf16_t* qbase = q    + ((b * 12 + h) * 1024 + i0) * 64;
  const bf16_t* kbase = kmat + (b * 12 + h) * 1024 * 64;
  const bf16_t* vbase = vT   + (b * 12 + h) * 64 * 1024;

  // q tile 16x64 preloaded as two A fragments (scale already folded into q)
  v16bf qa0 = load_frag(qbase + 0, 64);
  v16bf qa1 = load_frag(qbase + 32, 64);

  v8f o[4] = {};
  float rm[8], rl[8];
#pragma unroll
  for (int v = 0; v < 8; ++v) { rm[v] = -1e30f; rl[v] = 0.f; }

  bf16_t* myP = &ldsP[w][0];

  // preload K fragments for jt = 0
  v16bf kf0 = load_frag(kbase + 0, 64);
  v16bf kf1 = load_frag(kbase + 32, 64);
  v16bf kf2 = load_frag(kbase + 16 * 64, 64);
  v16bf kf3 = load_frag(kbase + 16 * 64 + 32, 64);

  for (int jt = 0; jt < 1024; jt += 32) {
    // ---- S = q @ k^T for 32 keys ----
    v8f s0 = {}, s1 = {};
    s0 = wmma_bf16(qa0, kf0, s0);
    s0 = wmma_bf16(qa1, kf1, s0);
    s1 = wmma_bf16(qa0, kf2, s1);
    s1 = wmma_bf16(qa1, kf3, s1);

    // ---- prefetch next K tile (flies under the softmax below) ----
    if (jt + 32 < 1024) {
      const bf16_t* kn = kbase + (jt + 32) * 64;
      kf0 = load_frag(kn + 0, 64);
      kf1 = load_frag(kn + 32, 64);
      kf2 = load_frag(kn + 16 * 64, 64);
      kf3 = load_frag(kn + 16 * 64 + 32, 64);
    }

    // ---- V fragments for this tile: independent of softmax, issue early ----
    v16bf vf[4];
#pragma unroll
    for (int c = 0; c < 4; ++c)
      vf[c] = load_frag(vbase + (c * 16) * 1024 + jt, 1024);

    // ---- elevation bias: clip(-alpha*relu((e_j-e_i)/1000), -10, 0) ----
    const float ej0 = eb[jt + col];
    const float ej1 = eb[jt + 16 + col];
#pragma unroll
    for (int v = 0; v < 8; ++v) {
      float d0 = (ej0 - e_i[v]) * 1e-3f;
      float p0 = -alpha * fmaxf(d0, 0.f);
      s0[v] += fminf(fmaxf(p0, -10.f), 0.f);
      float d1 = (ej1 - e_i[v]) * 1e-3f;
      float p1 = -alpha * fmaxf(d1, 0.f);
      s1[v] += fminf(fmaxf(p1, -10.f), 0.f);
    }

    // ---- online softmax over this 32-wide block ----
#pragma unroll
    for (int v = 0; v < 8; ++v) {
      float mx = fmaxf(s0[v], s1[v]);
#pragma unroll
      for (int m = 8; m >= 1; m >>= 1) mx = fmaxf(mx, __shfl_xor(mx, m, 16));
      const float m_new = fmaxf(rm[v], mx);
      const float sc = __expf(rm[v] - m_new);
      const float p0 = __expf(s0[v] - m_new);
      const float p1 = __expf(s1[v] - m_new);
      float ls = p0 + p1;
#pragma unroll
      for (int m = 8; m >= 1; m >>= 1) ls += __shfl_xor(ls, m, 16);
      rl[v] = rl[v] * sc + ls;
      rm[v] = m_new;
#pragma unroll
      for (int c = 0; c < 4; ++c) o[c][v] *= sc;
      const int mrow = v + half * 8;
      myP[mrow * 32 + col]      = (bf16_t)p0;   // C-layout -> row-major [i][j]
      myP[mrow * 32 + 16 + col] = (bf16_t)p1;
    }

    // ---- reload P as an A fragment (per-wave LDS, DS ops in-order) ----
    v16bf pa;
    {
      const int r  = lane & 15;
      const int kb = (lane >> 4) << 3;
      const bf16_t* rowp = myP + r * 32 + kb;
      v8bf lo = *(const v8bf*)(rowp);
      v8bf hi = *(const v8bf*)(rowp + 16);
#pragma unroll
      for (int i2 = 0; i2 < 8; ++i2) { pa[i2] = lo[i2]; pa[i2 + 8] = hi[i2]; }
    }

    // ---- O += P @ V  (B fragments from vT rows d, K = j) ----
#pragma unroll
    for (int c = 0; c < 4; ++c) o[c] = wmma_bf16(pa, vf[c], o[c]);
  }

  // ---- normalize and store heads output [b*N+i][h*64+d] as bf16 ----
  bf16_t* ob = att + (b * 1024 + i0) * 768 + h * 64;
#pragma unroll
  for (int v = 0; v < 8; ++v) {
    const float inv = 1.f / rl[v];
    const int i = v + half * 8;
#pragma unroll
    for (int c = 0; c < 4; ++c) {
      ob[i * 768 + c * 16 + col] = (bf16_t)(o[c][v] * inv);
    }
  }
}

// ---------------------------------------------------------------------------
// Kernel 3: projection GEMM  out = att @ Wproj^T + b   (M=4096, K=768, N=768)
// ---------------------------------------------------------------------------
__global__ void __launch_bounds__(256)
proj_gemm(const bf16_t* __restrict__ ab, const bf16_t* __restrict__ wb,
          const float* __restrict__ bias, float* __restrict__ out) {
  const int lane = threadIdx.x & 31;
  const int wave = blockIdx.x * (blockDim.x >> 5) + (threadIdx.x >> 5);
  const int row_t = wave / 12;       // 256 row tiles
  const int col_g = wave % 12;       // 12 col groups of 64
  const int i0 = row_t * 16;
  const int n0 = col_g * 64;

  v8f acc[4] = {};
  gemm_16x64_k768(ab + i0 * 768, wb + n0 * 768, acc);

  const int half = lane >> 4, col = lane & 15;
#pragma unroll
  for (int c = 0; c < 4; ++c) {
    const int n = n0 + c * 16 + col;
    const float bv = bias[n];
#pragma unroll
    for (int v = 0; v < 8; ++v) {
      const int i = i0 + v + half * 8;
      out[i * 768 + n] = acc[c][v] + bv;
    }
  }
}

// ---------------------------------------------------------------------------
// Launch
// ---------------------------------------------------------------------------
extern "C" void kernel_launch(void* const* d_in, const int* in_sizes, int n_in,
                              void* d_out, int out_size, void* d_ws, size_t ws_size,
                              hipStream_t stream) {
  (void)in_sizes; (void)n_in; (void)out_size; (void)ws_size;
  const float* x     = (const float*)d_in[0];
  const float* elev  = (const float*)d_in[1];
  const float* Wqkv  = (const float*)d_in[2];
  const float* Wproj = (const float*)d_in[3];
  const float* bproj = (const float*)d_in[4];
  const float* alpha = (const float*)d_in[5];

  char* ws = (char*)d_ws;
  bf16_t* xb     = (bf16_t*)(ws + 0);           //  4096*768  = 6,291,456 B
  bf16_t* wqkvb  = (bf16_t*)(ws + 6291456);     //  2304*768  = 3,538,944 B
  bf16_t* wprojb = (bf16_t*)(ws + 9830400);     //   768*768  = 1,179,648 B
  bf16_t* qbuf   = (bf16_t*)(ws + 11010048);    //  [4,12,1024,64] bf16
  bf16_t* kbuf   = (bf16_t*)(ws + 17301504);
  bf16_t* vTbuf  = (bf16_t*)(ws + 23592960);    //  [4,12,64,1024] bf16
  bf16_t* attbuf = (bf16_t*)(ws + 29884416);    //  [4096,768] bf16

  const int nx = 4 * 1024 * 768;
  const int nw = 2304 * 768;
  const int np = 768 * 768;
  cvt_f32_to_bf16<<<(nx + 255) / 256, 256, 0, stream>>>(x, xb, nx);
  cvt_f32_to_bf16<<<(nw + 255) / 256, 256, 0, stream>>>(Wqkv, wqkvb, nw);
  cvt_f32_to_bf16<<<(np + 255) / 256, 256, 0, stream>>>(Wproj, wprojb, np);

  // 256 row-tiles * 36 col-groups = 9216 waves / 8 per block
  qkv_gemm<<<1152, 256, 0, stream>>>(xb, wqkvb, qbuf, kbuf, vTbuf);
  // 4*12*64 = 3072 waves / 4 per block
  attn_kernel<<<768, 128, 0, stream>>>(qbuf, kbuf, vTbuf, elev, alpha, attbuf);
  // 256 * 12 = 3072 waves / 8 per block
  proj_gemm<<<384, 256, 0, stream>>>(attbuf, wprojb, bproj, (float*)d_out);
}
